// FNO_44100724195426
// MI455X (gfx1250) — matrix operand order repaired and checked by reading, not compiled, not run on hardware
//
#include <hip/hip_runtime.h>
#include <math.h>

#define NB   8
#define CIN  3
#define HID  64
#define HH   256
#define WW   256
#define PIX  (HH*WW)        // 65536
#define KY   16             // kept ky modes
#define NKX  32             // kept kx modes (16 pos + 16 neg)
#define PROJ 256
#define TWO_PI_OVER_256 0.024543692606170259f

typedef __attribute__((ext_vector_type(16))) _Float16 v16h;
typedef __attribute__((ext_vector_type(8)))  float    v8f;

union ABFrag { v16h v; unsigned int u[8]; };

#define WMMA_F16(a, b, c) \
    __builtin_amdgcn_wmma_f32_16x16x32_f16(false, (a), false, (b), (short)0, (c), false, false)

__device__ __forceinline__ float gelu_exact(float v) {
    return 0.5f * v * (1.0f + erff(v * 0.70710678118654752f));
}

// A-fragment K index (contiguous k = 16h + s, same order as b_kidx): pairs 2v(+1), lane-half selects K 0-15/16-31
__device__ __forceinline__ int a_kidx(int v, int hi) {
    return 2*v + hi*16;
}
// B-fragment K index (contiguous k = 16h + s, same order as a_kidx): pairs 2v(+1), lane-half selects K 0-15/16-31
__device__ __forceinline__ int b_kidx(int v, int hi) {
    return 2*v + hi*16;
}

// ---------------------------------------------------------------- lift (K=3)
__global__ __launch_bounds__(256)
void lift_kernel(const float* __restrict__ x, const float* __restrict__ lw,
                 const float* __restrict__ lb, _Float16* __restrict__ h16) {
    long long idx = (long long)blockIdx.x * blockDim.x + threadIdx.x;
    long long total = (long long)NB * HID * PIX;
    if (idx >= total) return;
    int p = (int)(idx % PIX);
    int o = (int)((idx / PIX) % HID);
    int b = (int)(idx / ((long long)PIX * HID));
    const float* xb = x + (long long)b * CIN * PIX + p;
    float v = lb[o];
    v += lw[o*CIN+0] * xb[0];
    v += lw[o*CIN+1] * xb[(long long)PIX];
    v += lw[o*CIN+2] * xb[2ll*PIX];
    h16[idx] = (_Float16)v;
}

// ---------------- forward truncated DFT along w as WMMA GEMM (ky=0..15)
// grid NB*HID*HH/64, block (32,4): 64 activation rows x K=256 x N=32 (ky,comp).
// Row slab staged with CDNA5 async global->LDS b128 copies (ASYNCcnt-tracked).
// Twiddles unscaled in f16 (2^-16 scale would be f16-denormal); norm in epilogue.
__global__ __launch_bounds__(128)
void fwd_w_wmma_kernel(const _Float16* __restrict__ h16, float* __restrict__ Xw) {
    __shared__ __align__(16) _Float16 rows[64*256];   // 32KB [r][w]
    __shared__ __align__(16) _Float16 twd[32*256];    // 16KB [n][w], n = 2*ky+comp
    const int tid = threadIdx.y * 32 + threadIdx.x;
    const long long rid0 = (long long)blockIdx.x * 64;

    __builtin_prefetch(h16 + (rid0 + 64) * WW, 0, 1);  // next block's rows

    // async stage: 32KB contiguous, 16B per lane per issue, 16 issues
    {
        unsigned lds0 = (unsigned)(uintptr_t)(&rows[0]);   // LDS offset = addr[31:0]
        const char* gsrc = (const char*)(h16 + rid0 * WW);
        #pragma unroll
        for (int it = 0; it < 16; ++it) {
            unsigned loff = lds0 + (unsigned)(tid * 16 + it * 2048);
            unsigned long long ga =
                (unsigned long long)(uintptr_t)(gsrc + tid * 16 + it * 2048);
            asm volatile("global_load_async_to_lds_b128 %0, %1, off"
                         :: "v"(loff), "v"(ga) : "memory");
        }
        asm volatile("s_wait_asynccnt 0x0" ::: "memory");
    }
    for (int j = tid; j < 32*256; j += 128) {
        int n = j >> 8, wv = j & 255;
        int ky = n >> 1, comp = n & 1;
        float ang = (float)((ky * wv) & 255) * TWO_PI_OVER_256;
        twd[j] = (_Float16)(comp ? -__sinf(ang) : __cosf(ang));   // e^{-i th}
    }
    __syncthreads();

    const int lane = threadIdx.x, lo = lane & 15, hi = lane >> 4;
    const int r0 = threadIdx.y * 16;
    const unsigned int* a32 = (const unsigned int*)rows;
    const unsigned int* b32 = (const unsigned int*)twd;

    v8f acc0 = {}, acc1 = {};
    #pragma unroll
    for (int kb = 0; kb < 256; kb += 32) {
        ABFrag a, b0, b1;
        #pragma unroll
        for (int v = 0; v < 8; ++v) {
            int ka = kb + a_kidx(v, hi);
            a.u[v]  = a32[((r0 + lo)*256 + ka) >> 1];
            int kB = kb + b_kidx(v, hi);
            b0.u[v] = b32[((0  + lo)*256 + kB) >> 1];
            b1.u[v] = b32[((16 + lo)*256 + kB) >> 1];
        }
        acc0 = WMMA_F16(a.v, b0.v, acc0);
        acc1 = WMMA_F16(a.v, b1.v, acc1);
    }
    const float sc = 1.0f / (float)(HH * WW);   // fft norm='forward'
    #pragma unroll
    for (int r = 0; r < 8; ++r) {
        long long row = rid0 + r0 + r + hi*8;
        Xw[row*32 + lo]      = acc0[r] * sc;
        Xw[row*32 + 16 + lo] = acc1[r] * sc;
    }
}

// ---------------------- forward truncated DFT along h (kx corners, complex)
__global__ __launch_bounds__(512)
void fwd_h_kernel(const float* __restrict__ Xw, float* __restrict__ XF) {
    __shared__ float slab[HH*KY*2];   // 32KB
    __shared__ float tab[256];
    const int bc = blockIdx.x;        // b*HID + c
    const float* src = Xw + (long long)bc * HH * KY * 2;
    const int t = threadIdx.x;
    for (int j = t; j < HH*KY*2; j += 512) slab[j] = src[j];
    for (int j = t; j < 256;     j += 512) tab[j] = __cosf((float)j * TWO_PI_OVER_256);
    __syncthreads();
    int j = t >> 4, ky = t & 15;
    int kx = (j < 16) ? j : (224 + j);          // 240..255 for corner 1
    float ar = 0.f, ai = 0.f;
    for (int hh = 0; hh < HH; ++hh) {
        int idx = (kx * hh) & 255;
        float c = tab[idx], s = tab[(idx + 192) & 255];
        float xr = slab[hh*32 + ky*2], xi = slab[hh*32 + ky*2 + 1];
        ar += xr*c + xi*s;                      // *(c - i s)
        ai += xi*c - xr*s;
    }
    float* dst = XF + (((long long)bc * NKX + j) * KY + ky) * 2;
    dst[0] = ar; dst[1] = ai;
}

// ------------------------------ per-mode 64x64 complex channel mix (einsum)
__global__ __launch_bounds__(64)
void mode_mul_kernel(const float* __restrict__ XF, const float* __restrict__ sw,
                     float* __restrict__ YF) {
    __shared__ float xin[HID*2];
    int blk = blockIdx.x;
    int ky = blk & 15;
    int j  = (blk >> 4) & 31;
    int b  = blk >> 9;
    int corner = (j < 16) ? 0 : 1;
    int kxl    = (j < 16) ? j : (j - 16);
    int o = threadIdx.x;
    {
        const float* p = XF + (((long long)(b*HID + o) * NKX + j) * KY + ky) * 2;
        xin[o*2] = p[0]; xin[o*2+1] = p[1];
    }
    __syncthreads();
    float ar = 0.f, ai = 0.f;
    long long base = ((((long long)corner*HID*HID + o)*16 + kxl)*16 + ky) * 2;
    const long long istr = (long long)HID*16*16*2;
    for (int i = 0; i < HID; ++i) {
        float wr = sw[base + (long long)i*istr];
        float wi = sw[base + (long long)i*istr + 1];
        float xr = xin[i*2], xi = xin[i*2+1];
        ar += xr*wr - xi*wi;
        ai += xr*wi + xi*wr;
    }
    float* dst = YF + (((long long)(b*HID + o) * NKX + j) * KY + ky) * 2;
    dst[0] = ar; dst[1] = ai;
}

// ---------------------------------- inverse DFT over kx (forward-norm: x1)
__global__ __launch_bounds__(256)
void inv_h_kernel(const float* __restrict__ YF, float* __restrict__ Z) {
    __shared__ float slab[NKX*KY*2];
    __shared__ float tab[256];
    int bo = blockIdx.x;              // b*HID + o
    const float* src = YF + (long long)bo * NKX * KY * 2;
    int t = threadIdx.x;
    for (int j = t; j < NKX*KY*2; j += 256) slab[j] = src[j];
    tab[t] = __cosf((float)t * TWO_PI_OVER_256);
    __syncthreads();
    int hh = t;
    float zr[KY], zi[KY];
    #pragma unroll
    for (int k = 0; k < KY; ++k) { zr[k] = 0.f; zi[k] = 0.f; }
    for (int j = 0; j < NKX; ++j) {
        int kx = (j < 16) ? j : (224 + j);
        int idx = (kx * hh) & 255;
        float c = tab[idx], s = tab[(idx + 192) & 255];   // e^{+i th}
        #pragma unroll
        for (int k = 0; k < KY; ++k) {
            float yr = slab[(j*KY + k)*2], yi = slab[(j*KY + k)*2 + 1];
            zr[k] += yr*c - yi*s;
            zi[k] += yr*s + yi*c;
        }
    }
    float* dst = Z + ((long long)bo * HH + hh) * KY * 2;
    #pragma unroll
    for (int k = 0; k < KY; ++k) { dst[k*2] = zr[k]; dst[k*2+1] = zi[k]; }
}

// ----- FUSED: skip 1x1 conv GEMM + inverse-ky GEMM + bias + exact GELU
// grid (WW/128, HH, NB), block (32,4). For a fixed (b, h-line, 128-w tile),
// D[o][w] = W_skip[o][i] x h16[i][w]   (K=64, 2 WMMAs)
//         + Zmod[o][kc]  x T[kc][w]    (K=32, 1 WMMA; Hermitian 2x + DC folded)
// then + bias, GELU, stored in place into h16. Each block stages every input
// channel of its own region into LDS before writing it -> in-place is safe.
__global__ __launch_bounds__(128)
void spec_skip_fused_kernel(const float* __restrict__ Z,
                            const float* __restrict__ skw,   // [64][64] (o,i)
                            const float* __restrict__ sb,    // [64]
                            _Float16* __restrict__ h16, int apply_gelu) {
    __shared__ _Float16 tA[HID*HID];    // 8KB  weights [o][i]
    __shared__ _Float16 tB[128*HID];    // 16KB activations [p][i]
    __shared__ _Float16 zt[HID*32];     // 4KB  Z modes [o][kc], kc = 2k+comp
    __shared__ _Float16 twd[128*32];    // 8KB  twiddles [p][kc]
    __shared__ float sbias[HID];
    const int tid = threadIdx.y * 32 + threadIdx.x;
    const int p0 = blockIdx.x * 128;
    const int hh = blockIdx.y;
    const int b  = blockIdx.z;

    for (int j = tid; j < HID*HID; j += 128) tA[j] = (_Float16)skw[j];
    _Float16* hb = h16 + (long long)b*HID*PIX + (long long)hh*WW + p0;
    for (int j = tid; j < 128*HID; j += 128) {
        int i = j >> 7, p = j & 127;
        tB[p*HID + i] = hb[(long long)i*PIX + p];
    }
    const float* zrow = Z + ((long long)b*HID*HH + hh) * 32;
    for (int j = tid; j < HID*32; j += 128) {
        int o = j >> 5, kc = j & 31;
        zt[j] = (_Float16)zrow[(long long)o*HH*32 + kc];
    }
    for (int j = tid; j < 128*32; j += 128) {
        int p = j >> 5, kc = j & 31;
        int k = kc >> 1, comp = kc & 1;
        float ang = (float)((k * (p0 + p)) & 255) * TWO_PI_OVER_256;
        float v = (comp == 0) ? (k == 0 ? 1.0f : 2.0f) * __cosf(ang)
                              : (k == 0 ? 0.0f : -2.0f * __sinf(ang));
        twd[j] = (_Float16)v;
    }
    if (tid < HID) sbias[tid] = sb[tid];
    __syncthreads();

    const int lane = threadIdx.x, lo = lane & 15, hi = lane >> 4;
    const int o0 = threadIdx.y * 16;
    const unsigned int* a32 = (const unsigned int*)tA;
    const unsigned int* b32 = (const unsigned int*)tB;
    const unsigned int* z32 = (const unsigned int*)zt;
    const unsigned int* t32 = (const unsigned int*)twd;

    ABFrag a0, a1, az;                   // held across all 8 w-tiles
    #pragma unroll
    for (int v = 0; v < 8; ++v) {
        int ka = a_kidx(v, hi);
        a0.u[v] = a32[((o0 + lo)*HID + ka) >> 1];
        a1.u[v] = a32[((o0 + lo)*HID + 32 + ka) >> 1];
        az.u[v] = z32[((o0 + lo)*32 + ka) >> 1];
    }
    #pragma unroll
    for (int np = 0; np < 8; ++np) {
        int p = np * 16;
        ABFrag b0, b1, bw;
        #pragma unroll
        for (int v = 0; v < 8; ++v) {
            int kB = b_kidx(v, hi);
            b0.u[v] = b32[((p + lo)*HID + kB) >> 1];
            b1.u[v] = b32[((p + lo)*HID + 32 + kB) >> 1];
            bw.u[v] = t32[((p + lo)*32 + kB) >> 1];
        }
        v8f acc = {};
        acc = WMMA_F16(a0.v, b0.v, acc);   // skip, K 0..31
        acc = WMMA_F16(a1.v, b1.v, acc);   // skip, K 32..63
        acc = WMMA_F16(az.v, bw.v, acc);   // spectral inverse over ky
        #pragma unroll
        for (int r = 0; r < 8; ++r) {
            int o = o0 + r + hi*8;
            float val = acc[r] + sbias[o];
            if (apply_gelu) val = gelu_exact(val);
            hb[(long long)o*PIX + p + lo] = (_Float16)val;
        }
    }
}

// -------------- fused projection: WMMA 64->256, GELU, dot 256->1 per pixel
// grid (PIX/128,1,NB), block (32,16): W1 staged once, 8 pixel tiles per block.
__global__ __launch_bounds__(512)
void proj_wmma_kernel(const _Float16* __restrict__ h16,
                      const float* __restrict__ w1, const float* __restrict__ b1,
                      const float* __restrict__ w2, const float* __restrict__ b2,
                      float* __restrict__ out) {
    __shared__ _Float16 tW[PROJ*HID];     // 32KB [o][i]
    __shared__ _Float16 tB[128*HID];      // 16KB [p][i]
    __shared__ float w2s[PROJ];
    __shared__ float b1s[PROJ];
    __shared__ float partials[16][16];    // [wave][p]
    const int tid = threadIdx.y * 32 + threadIdx.x;
    const int b = blockIdx.z, p0 = blockIdx.x * 128;
    for (int j = tid; j < PROJ*HID; j += 512) tW[j] = (_Float16)w1[j];
    const _Float16* hb = h16 + (long long)b * HID * PIX;
    for (int j = tid; j < 128*HID; j += 512) {
        int i = j >> 7, p = j & 127;
        tB[p*HID + i] = hb[(long long)i * PIX + p0 + p];
    }
    if (tid < PROJ) { w2s[tid] = w2[tid]; b1s[tid] = b1[tid]; }
    __syncthreads();

    const int lane = threadIdx.x, lo = lane & 15, hi = lane >> 4;
    const int m0 = threadIdx.y * 16;
    const unsigned int* a32 = (const unsigned int*)tW;
    const unsigned int* bb  = (const unsigned int*)tB;

    ABFrag a0, a1;
    #pragma unroll
    for (int v = 0; v < 8; ++v) {
        int ka = a_kidx(v, hi);
        a0.u[v] = a32[((m0 + lo)*HID + ka) >> 1];
        a1.u[v] = a32[((m0 + lo)*HID + 32 + ka) >> 1];
    }
    for (int np = 0; np < 8; ++np) {
        int p = np * 16;
        ABFrag bf0, bf1;
        #pragma unroll
        for (int v = 0; v < 8; ++v) {
            int kB = b_kidx(v, hi);
            bf0.u[v] = bb[((p + lo)*HID + kB) >> 1];
            bf1.u[v] = bb[((p + lo)*HID + 32 + kB) >> 1];
        }
        v8f acc = {};
        acc = WMMA_F16(a0.v, bf0.v, acc);
        acc = WMMA_F16(a1.v, bf1.v, acc);
        float part = 0.f;
        #pragma unroll
        for (int r = 0; r < 8; ++r) {
            int o = m0 + r + hi*8;
            part += gelu_exact(acc[r] + b1s[o]) * w2s[o];
        }
        part += __shfl_xor(part, 16, 32);   // lanes L, L+16 share a pixel
        __syncthreads();                     // protect partials reuse
        if (hi == 0) partials[threadIdx.y][lo] = part;
        __syncthreads();
        if (tid < 16) {
            float tot = 0.f;
            #pragma unroll
            for (int wd = 0; wd < 16; ++wd) tot += partials[wd][tid];
            out[(long long)b * PIX + p0 + p + tid] = tot + b2[0];
        }
    }
}

// ---------------------------------------------------------------- launcher
extern "C" void kernel_launch(void* const* d_in, const int* in_sizes, int n_in,
                              void* d_out, int out_size, void* d_ws, size_t ws_size,
                              hipStream_t stream) {
    const float* x      = (const float*)d_in[0];
    const float* lift_w = (const float*)d_in[1];
    const float* lift_b = (const float*)d_in[2];
    const float* spec_w = (const float*)d_in[3];
    const float* spec_b = (const float*)d_in[4];
    const float* skip_w = (const float*)d_in[5];
    const float* p1w    = (const float*)d_in[6];
    const float* p1b    = (const float*)d_in[7];
    const float* p2w    = (const float*)d_in[8];
    const float* p2b    = (const float*)d_in[9];
    float* out = (float*)d_out;

    char* ws = (char*)d_ws;
    size_t off = 0;
    _Float16* h16 = (_Float16*)(ws + off); off += (size_t)NB*HID*PIX*2;        // 64MB
    float*    Xw  = (float*)(ws + off);    off += (size_t)NB*HID*HH*KY*2*4;    // 16MB
    float*    XF  = (float*)(ws + off);    off += (size_t)NB*HID*NKX*KY*2*4;   // 2MB
    float*    YF  = (float*)(ws + off);    off += (size_t)NB*HID*NKX*KY*2*4;   // 2MB
    float*    Z   = (float*)(ws + off);    off += (size_t)NB*HID*HH*KY*2*4;    // 16MB

    long long total = (long long)NB*HID*PIX;
    lift_kernel<<<(unsigned)((total+255)/256), 256, 0, stream>>>(x, lift_w, lift_b, h16);

    const int nrows = NB*HID*HH;            // 131072 activation rows
    for (int l = 0; l < 4; ++l) {
        fwd_w_wmma_kernel<<<nrows/64, dim3(32,4), 0, stream>>>(h16, Xw);
        fwd_h_kernel<<<NB*HID, 512, 0, stream>>>(Xw, XF);
        mode_mul_kernel<<<NB*NKX*KY, HID, 0, stream>>>(
            XF, spec_w + (size_t)l*2*HID*HID*16*16*2, YF);
        inv_h_kernel<<<NB*HID, HH, 0, stream>>>(YF, Z);
        spec_skip_fused_kernel<<<dim3(WW/128, HH, NB), dim3(32,4), 0, stream>>>(
            Z, skip_w + (size_t)l*HID*HID, spec_b + l*HID, h16, (l < 3) ? 1 : 0);
    }
    proj_wmma_kernel<<<dim3(PIX/128,1,NB), dim3(32,16), 0, stream>>>(
        h16, p1w, p1b, p2w, p2b, out);
}
